// MultiHeadAttention_41472204210594
// MI455X (gfx1250) — matrix operand order repaired and checked by reading, not compiled, not run on hardware
//
#include <hip/hip_runtime.h>

typedef _Float16 v16h __attribute__((ext_vector_type(16)));
typedef _Float16 v8h  __attribute__((ext_vector_type(8)));
typedef float    v8f  __attribute__((ext_vector_type(8)));

// ---------------------------------------------------------------------------
// Constants for this problem (B=2, S=2048, D=1024, H=16, HD=64)
// ---------------------------------------------------------------------------
#define CB   2
#define CS   2048
#define CD   1024
#define CH   16
#define CHD  64
#define CN   (CB * CS)        // 4096 rows

static __device__ __forceinline__ v8f wmma16(v16h a, v16h b, v8f c) {
  // D = A(16x32 f16) x B(32x16 f16) + C(16x16 f32)
  return __builtin_amdgcn_wmma_f32_16x16x32_f16(
      /*neg_a=*/false, a, /*neg_b=*/false, b,
      /*c_mod=*/(short)0, c, /*reuse_a=*/false, /*reuse_b=*/false);
}

static __device__ __forceinline__ v16h ldfrag(const _Float16* p) {
  return *(const v16h*)p;
}

// ---------------------------------------------------------------------------
// Fragment packing kernels.
// A-frag layout (16x32, f16): half index e, lane:
//   M = lane & 15,  K = e + (e>=8 ? 8 : 0) + (lane>=16 ? 8 : 0)
// B-frag layout (32x16, f16):
//   N = lane & 15,  K = e + (e>=8 ? 8 : 0) + (lane>=16 ? 8 : 0)
// Packed storage: frag t -> dst[(t*32 + lane)*16 + e]
// ---------------------------------------------------------------------------

// x (f32 row-major [rows x K]) -> A frags, tile order t = mTile*KT + kTile
__global__ void pack_a_f32(const float* __restrict__ src, _Float16* __restrict__ dst,
                           int KT, int ld, int total) {
  int idx = blockIdx.x * blockDim.x + threadIdx.x;
  if (idx >= total) return;
  int e = idx & 15;
  int lane = (idx >> 4) & 31;
  int t = idx >> 9;
  int kTile = t % KT;
  int mTile = t / KT;
  int m = mTile * 16 + (lane & 15);
  int k = kTile * 32 + e + ((e >= 8) ? 8 : 0) + ((lane >= 16) ? 8 : 0);
  dst[idx] = (_Float16)src[(size_t)m * ld + k];
}

// ctx (f16 row-major [rows x K]) -> A frags (same mapping as above)
__global__ void pack_a_f16(const _Float16* __restrict__ src, _Float16* __restrict__ dst,
                           int KT, int ld, int total) {
  int idx = blockIdx.x * blockDim.x + threadIdx.x;
  if (idx >= total) return;
  int e = idx & 15;
  int lane = (idx >> 4) & 31;
  int t = idx >> 9;
  int kTile = t % KT;
  int mTile = t / KT;
  int m = mTile * 16 + (lane & 15);
  int k = kTile * 32 + e + ((e >= 8) ? 8 : 0) + ((lane >= 16) ? 8 : 0);
  dst[idx] = src[(size_t)m * ld + k];
}

// W (f32 row-major [K x Mcols]) -> B frags, tile order t = nTile*KT + kTile
__global__ void pack_b_f32(const float* __restrict__ src, _Float16* __restrict__ dst,
                           int KT, int ld, int total) {
  int idx = blockIdx.x * blockDim.x + threadIdx.x;
  if (idx >= total) return;
  int e = idx & 15;
  int lane = (idx >> 4) & 31;
  int t = idx >> 9;
  int kTile = t % KT;
  int nTile = t / KT;
  int n = nTile * 16 + (lane & 15);
  int k = kTile * 32 + e + ((e >= 8) ? 8 : 0) + ((lane >= 16) ? 8 : 0);
  dst[idx] = (_Float16)src[(size_t)k * ld + n];
}

// q [N x D] f16 -> per-head A frags, t = ((bh*128 + mTile)*2 + kTile)
// Folds the 1/sqrt(HD)=0.125 attention scale (exact in f16: power of two).
__global__ void pack_qA(const _Float16* __restrict__ src, _Float16* __restrict__ dst,
                        int total) {
  int idx = blockIdx.x * blockDim.x + threadIdx.x;
  if (idx >= total) return;
  int e = idx & 15;
  int lane = (idx >> 4) & 31;
  int t = idx >> 9;
  int kTile = t & 1;
  int mTile = (t >> 1) & 127;
  int bh = t >> 8;
  int b = bh >> 4, h = bh & 15;
  int s = mTile * 16 + (lane & 15);
  int hd = kTile * 32 + e + ((e >= 8) ? 8 : 0) + ((lane >= 16) ? 8 : 0);
  float v = (float)src[((size_t)(b * CS + s)) * CD + h * CHD + hd];
  dst[idx] = (_Float16)(v * 0.125f);
}

// k [N x D] f16 -> per-head B frags of K^T, t = ((bh*128 + nTile)*2 + kTile)
// B[kk_hd, n_key] = k[n_key, kk_hd]
__global__ void pack_kB(const _Float16* __restrict__ src, _Float16* __restrict__ dst,
                        int total) {
  int idx = blockIdx.x * blockDim.x + threadIdx.x;
  if (idx >= total) return;
  int e = idx & 15;
  int lane = (idx >> 4) & 31;
  int t = idx >> 9;
  int kTile = t & 1;
  int nTile = (t >> 1) & 127;
  int bh = t >> 8;
  int b = bh >> 4, h = bh & 15;
  int key = nTile * 16 + (lane & 15);
  int hd = kTile * 32 + e + ((e >= 8) ? 8 : 0) + ((lane >= 16) ? 8 : 0);
  dst[idx] = src[((size_t)(b * CS + key)) * CD + h * CHD + hd];
}

// v [N x D] f16 -> per-head B frags, t = ((bh*64 + kTile)*4 + nTile)
// B[key, hd] = v[key, hd]
__global__ void pack_vB(const _Float16* __restrict__ src, _Float16* __restrict__ dst,
                        int total) {
  int idx = blockIdx.x * blockDim.x + threadIdx.x;
  if (idx >= total) return;
  int e = idx & 15;
  int lane = (idx >> 4) & 31;
  int t = idx >> 9;
  int nTile = t & 3;
  int kTile = (t >> 2) & 63;
  int bh = t >> 8;
  int b = bh >> 4, h = bh & 15;
  int key = kTile * 32 + e + ((e >= 8) ? 8 : 0) + ((lane >= 16) ? 8 : 0);
  int hd = nTile * 16 + (lane & 15);
  dst[idx] = src[((size_t)(b * CS + key)) * CD + h * CHD + hd];
}

// ---------------------------------------------------------------------------
// GEMM: out = relu(A @ W + bias). A pre-packed A-frags, W pre-packed B-frags.
// Wave tile: 32 rows x 64 cols (2x4 accumulators). Explicit 2x-unrolled
// ping-pong double buffer: even steps consume buf0/load buf1, odd steps the
// reverse -- no register rotation copies, 16 static WMMAs in the loop body.
// ---------------------------------------------------------------------------
__global__ __launch_bounds__(128) void gemm_wmma_kernel(
    const _Float16* __restrict__ Ap, const _Float16* __restrict__ Bp,
    const float* __restrict__ bias, _Float16* __restrict__ outH,
    float* __restrict__ outF, int Mcols, int KT) {
  const int lane = threadIdx.x & 31;
  const int wave = (blockIdx.x * blockDim.x + threadIdx.x) >> 5;
  const int colGroups = Mcols >> 6;
  const int rowTile = wave / colGroups;
  const int colTile = wave % colGroups;
  const int mT0 = rowTile * 2;
  const int nT0 = colTile * 4;

  // Base pointers; consecutive kt is a +512-half stride.
  const _Float16* a0p = Ap + (size_t)mT0 * KT * 512 + lane * 16;
  const _Float16* a1p = Ap + (size_t)(mT0 + 1) * KT * 512 + lane * 16;
  const _Float16* bp[4];
#pragma unroll
  for (int j = 0; j < 4; ++j) bp[j] = Bp + (size_t)(nT0 + j) * KT * 512 + lane * 16;

  v8f acc[2][4] = {};

  v16h a0[2], a1[2], bf[2][4];
  // Prime buffer 0 with kt = 0.
  a0[0] = ldfrag(a0p);
  a1[0] = ldfrag(a1p);
#pragma unroll
  for (int j = 0; j < 4; ++j) bf[0][j] = ldfrag(bp[j]);

  for (int kt = 0; kt < KT; kt += 2) {            // KT is even
    // Load buffer 1 with kt+1 (always in range).
    {
      const size_t o = (size_t)(kt + 1) * 512;
      a0[1] = ldfrag(a0p + o);
      a1[1] = ldfrag(a1p + o);
#pragma unroll
      for (int j = 0; j < 4; ++j) bf[1][j] = ldfrag(bp[j] + o);
    }
    {
      const int ktp = (kt + 3 < KT) ? kt + 3 : KT - 1;
      __builtin_prefetch(a0p + (size_t)ktp * 512, 0, 1);
      __builtin_prefetch(bp[0] + (size_t)ktp * 512, 0, 1);
    }
    // Consume buffer 0 (kt).
#pragma unroll
    for (int j = 0; j < 4; ++j) {
      acc[0][j] = wmma16(a0[0], bf[0][j], acc[0][j]);
      acc[1][j] = wmma16(a1[0], bf[0][j], acc[1][j]);
    }
    // Load buffer 0 with kt+2 (clamped on the last trip).
    {
      const int k2 = (kt + 2 < KT) ? kt + 2 : KT - 1;
      const size_t o = (size_t)k2 * 512;
      a0[0] = ldfrag(a0p + o);
      a1[0] = ldfrag(a1p + o);
#pragma unroll
      for (int j = 0; j < 4; ++j) bf[0][j] = ldfrag(bp[j] + o);
    }
    // Consume buffer 1 (kt+1).
#pragma unroll
    for (int j = 0; j < 4; ++j) {
      acc[0][j] = wmma16(a0[1], bf[1][j], acc[0][j]);
      acc[1][j] = wmma16(a1[1], bf[1][j], acc[1][j]);
    }
  }

  const int hiHalf = (lane >= 16) ? 8 : 0;
  const int c = lane & 15;
#pragma unroll
  for (int i = 0; i < 2; ++i) {
#pragma unroll
    for (int j = 0; j < 4; ++j) {
      int colB = colTile * 64 + j * 16 + c;
      float bia = bias[colB];
#pragma unroll
      for (int r = 0; r < 8; ++r) {
        int row = rowTile * 32 + i * 16 + r + hiHalf;
        float v = acc[i][j][r] + bia;
        v = fmaxf(v, 0.0f);
        size_t o = (size_t)row * Mcols + colB;
        if (outH) outH[o] = (_Float16)v;
        else      outF[o] = v;
      }
    }
  }
}

// ---------------------------------------------------------------------------
// Flash attention: one wave per (b, h, 16-query tile). Streams 32 keys/iter:
// 4 WMMAs for Q@K^T (K-dim = HD = 64), online softmax via lane-half shuffles,
// P tile relaid through wave-private LDS into A-frag form, 4 WMMAs for P@V.
// 2x-unrolled ping-pong buffering of K/V fragments (no rotation copies).
// ---------------------------------------------------------------------------
__global__ __launch_bounds__(128) void flash_attn_kernel(
    const _Float16* __restrict__ qA, const _Float16* __restrict__ kB,
    const _Float16* __restrict__ vB, _Float16* __restrict__ ctx) {
  __shared__ _Float16 smem[4 * 512];   // 1KB (16x32 f16) per wave
  const int lane = threadIdx.x & 31;
  const int wslot = threadIdx.x >> 5;
  const int wave = (blockIdx.x * blockDim.x + threadIdx.x) >> 5;
  const int qt = wave & 127;
  const int bh = wave >> 7;
  const int b = bh >> 4, h = bh & 15;

  const _Float16* qbase = qA + (((size_t)bh * 128 + qt) * 2) * 512 + lane * 16;
  const v16h qa0 = ldfrag(qbase);
  const v16h qa1 = ldfrag(qbase + 512);

  v8f acc[4] = {};
  v8f mrun, lrun;
#pragma unroll
  for (int r = 0; r < 8; ++r) { mrun[r] = -3.0e38f; lrun[r] = 0.0f; }

  const int wbase = wslot * 512;
  const int c = lane & 15;
  const int hiHalf = (lane >= 16) ? 8 : 0;

  const _Float16* kbBase = kB + (size_t)bh * 256 * 512 + lane * 16; // frag (nT,kT): +((nT*2+kT)*512)
  const _Float16* vbBase = vB + (size_t)bh * 256 * 512 + lane * 16; // frag (kT,nT): +((kT*4+nT)*512)

  // One 32-key tile: scores -> online softmax -> P@V accumulate.
  auto tile = [&](const v16h* ck, const v16h* cv) {
    v8f z = {};
    v8f e0 = wmma16(qa1, ck[1], wmma16(qa0, ck[0], z));
    v8f e1 = wmma16(qa1, ck[3], wmma16(qa0, ck[2], z));

    v8f corr;
#pragma unroll
    for (int r = 0; r < 8; ++r) {
      float mx = fmaxf(e0[r], e1[r]);
      mx = fmaxf(mx, __shfl_xor(mx, 1));
      mx = fmaxf(mx, __shfl_xor(mx, 2));
      mx = fmaxf(mx, __shfl_xor(mx, 4));
      mx = fmaxf(mx, __shfl_xor(mx, 8));
      float mn = fmaxf(mrun[r], mx);
      float cr = __expf(mrun[r] - mn);
      mrun[r] = mn;
      float p0 = __expf(e0[r] - mn);
      float p1 = __expf(e1[r] - mn);
      e0[r] = p0; e1[r] = p1;
      float rs = p0 + p1;
      rs += __shfl_xor(rs, 1);
      rs += __shfl_xor(rs, 2);
      rs += __shfl_xor(rs, 4);
      rs += __shfl_xor(rs, 8);
      lrun[r] = lrun[r] * cr + rs;
      corr[r] = cr;
    }
#pragma unroll
    for (int j = 0; j < 4; ++j)
#pragma unroll
      for (int r = 0; r < 8; ++r) acc[j][r] *= corr[r];

    // D-layout P -> LDS (row-major 16x32) -> A-frag reload.
    __syncthreads();
#pragma unroll
    for (int r = 0; r < 8; ++r) {
      int row = r + hiHalf;
      smem[wbase + row * 32 + c]      = (_Float16)e0[r];
      smem[wbase + row * 32 + 16 + c] = (_Float16)e1[r];
    }
    __syncthreads();
    const _Float16* pp = smem + wbase + (lane & 15) * 32 + hiHalf;
    v8h lo = *(const v8h*)(pp);        // K = hiHalf .. hiHalf+7
    v8h hi = *(const v8h*)(pp + 16);   // K = 16+hiHalf .. 23+hiHalf
    v16h pa;
#pragma unroll
    for (int t = 0; t < 8; ++t) { pa[t] = lo[t]; pa[t + 8] = hi[t]; }

#pragma unroll
    for (int j = 0; j < 4; ++j) acc[j] = wmma16(pa, cv[j], acc[j]);
  };

  v16h ck[2][4], cv[2][4];
  // Prime buffer 0 with kp = 0.
#pragma unroll
  for (int f = 0; f < 4; ++f) {
    ck[0][f] = ldfrag(kbBase + (size_t)f * 512);
    cv[0][f] = ldfrag(vbBase + (size_t)f * 512);
  }

  for (int kp = 0; kp < 64; kp += 2) {            // 64 tiles, even count
    // Load buffer 1 with kp+1 (always in range).
#pragma unroll
    for (int f = 0; f < 4; ++f) {
      ck[1][f] = ldfrag(kbBase + ((size_t)(kp + 1) * 4 + f) * 512);
      cv[1][f] = ldfrag(vbBase + ((size_t)(kp + 1) * 4 + f) * 512);
    }
    __builtin_prefetch(kbBase + ((size_t)((kp + 3 < 64) ? kp + 3 : 63) * 4) * 512, 0, 1);

    tile(ck[0], cv[0]);                           // consume kp

    // Load buffer 0 with kp+2 (clamped on the last trip).
    {
      const int k2 = (kp + 2 < 64) ? kp + 2 : 63;
#pragma unroll
      for (int f = 0; f < 4; ++f) {
        ck[0][f] = ldfrag(kbBase + ((size_t)k2 * 4 + f) * 512);
        cv[0][f] = ldfrag(vbBase + ((size_t)k2 * 4 + f) * 512);
      }
    }

    tile(ck[1], cv[1]);                           // consume kp+1
  }

  // Normalize and write ctx [B,S,H,HD] row-major f16.
#pragma unroll
  for (int r = 0; r < 8; ++r) {
    float inv = 1.0f / lrun[r];
    int row = qt * 16 + r + hiHalf;
    size_t base = ((size_t)(b * CS + row)) * CD + h * CHD + c;
#pragma unroll
    for (int j = 0; j < 4; ++j)
      ctx[base + j * 16] = (_Float16)(acc[j][r] * inv);
  }
}

// ---------------------------------------------------------------------------
// Host orchestration
// ---------------------------------------------------------------------------
extern "C" void kernel_launch(void* const* d_in, const int* in_sizes, int n_in,
                              void* d_out, int out_size, void* d_ws, size_t ws_size,
                              hipStream_t stream) {
  const float* x  = (const float*)d_in[0];
  const float* Wq = (const float*)d_in[1];
  const float* bq = (const float*)d_in[2];
  const float* Wk = (const float*)d_in[3];
  const float* bk = (const float*)d_in[4];
  const float* Wv = (const float*)d_in[5];
  const float* bv = (const float*)d_in[6];
  const float* Wo = (const float*)d_in[7];
  const float* bo = (const float*)d_in[8];
  float* out = (float*)d_out;

  char* ws = (char*)d_ws;
  _Float16* xA   = (_Float16*)(ws + 0);          //  8 MiB: x A-frags
  _Float16* WqB  = (_Float16*)(ws + (8  << 20)); //  2 MiB each
  _Float16* WkB  = (_Float16*)(ws + (10 << 20));
  _Float16* WvB  = (_Float16*)(ws + (12 << 20));
  _Float16* WoB  = (_Float16*)(ws + (14 << 20));
  _Float16* qh   = (_Float16*)(ws + (16 << 20)); //  8 MiB each, row-major [N,D]
  _Float16* kh   = (_Float16*)(ws + (24 << 20));
  _Float16* vh   = (_Float16*)(ws + (32 << 20));
  _Float16* qAf  = (_Float16*)(ws + (40 << 20)); //  8 MiB packed per-head frags
  _Float16* kBf  = (_Float16*)(ws + (48 << 20));
  _Float16* vBf  = (_Float16*)(ws + (56 << 20));
  _Float16* ctxh = (_Float16*)(ws + (64 << 20)); //  8 MiB row-major [N,D]
  _Float16* ctxA = (_Float16*)(ws + (72 << 20)); //  8 MiB A-frags

  const int KT = CD / 32;                 // 32 k-tiles
  const int totA  = (CN / 16) * KT * 512; // 4,194,304 halves
  const int totB  = (CD / 16) * KT * 512; // 1,048,576 halves
  const int totPH = 32 * 256 * 512;       // per-head packs: 4,194,304 halves

  // 1) pack activations + weights into WMMA fragment layouts
  pack_a_f32<<<totA / 256, 256, 0, stream>>>(x, xA, KT, CD, totA);
  pack_b_f32<<<totB / 256, 256, 0, stream>>>(Wq, WqB, KT, CD, totB);
  pack_b_f32<<<totB / 256, 256, 0, stream>>>(Wk, WkB, KT, CD, totB);
  pack_b_f32<<<totB / 256, 256, 0, stream>>>(Wv, WvB, KT, CD, totB);
  pack_b_f32<<<totB / 256, 256, 0, stream>>>(Wo, WoB, KT, CD, totB);

  // 2) q/k/v = relu(x@W + b), f16 outputs     (2048 waves each)
  gemm_wmma_kernel<<<512, 128, 0, stream>>>(xA, WqB, bq, qh, nullptr, CD, KT);
  gemm_wmma_kernel<<<512, 128, 0, stream>>>(xA, WkB, bk, kh, nullptr, CD, KT);
  gemm_wmma_kernel<<<512, 128, 0, stream>>>(xA, WvB, bv, vh, nullptr, CD, KT);

  // 3) repack q/k/v per head into attention fragment layouts
  pack_qA<<<totPH / 256, 256, 0, stream>>>(qh, qAf, totPH);
  pack_kB<<<totPH / 256, 256, 0, stream>>>(kh, kBf, totPH);
  pack_vB<<<totPH / 256, 256, 0, stream>>>(vh, vBf, totPH);

  // 4) flash attention (4096 waves: B*H*S/16)
  flash_attn_kernel<<<1024, 128, 0, stream>>>(qAf, kBf, vBf, ctxh);

  // 5) out = relu(ctx@Wo + bo), f32 output
  pack_a_f16<<<totA / 256, 256, 0, stream>>>(ctxh, ctxA, KT, CD, totA);
  gemm_wmma_kernel<<<512, 128, 0, stream>>>(ctxA, WoB, bo, nullptr, out, CD, KT);
}
